// LinearTransformerXL_5033701670965
// MI455X (gfx1250) — matrix operand order repaired and checked by reading, not compiled, not run on hardware
//
#include <hip/hip_runtime.h>

// ---------------------------------------------------------------------------
// Types for CDNA5 WMMA (wave32, 16x16x32 f16 -> f32)
// ---------------------------------------------------------------------------
typedef __attribute__((ext_vector_type(16))) _Float16 v16h;
typedef __attribute__((ext_vector_type(8)))  _Float16 v8h;
typedef __attribute__((ext_vector_type(8)))  float    v8f;
typedef __attribute__((ext_vector_type(4)))  int      v4i;

__device__ __forceinline__ v8f wmma16(v16h a, v16h b, v8f c) {
    // (neg_a, A, neg_b, B, c_mod, C, reuse_a, reuse_b)
    return __builtin_amdgcn_wmma_f32_16x16x32_f16(false, a, false, b,
                                                  (short)0, c, false, false);
}

// Assemble a 16-half fragment from two 16B-aligned locations (LDS or global).
__device__ __forceinline__ v16h frag16(const _Float16* lo, const _Float16* hi) {
    v8h a = *reinterpret_cast<const v8h*>(lo);
    v8h b = *reinterpret_cast<const v8h*>(hi);
    return __builtin_shufflevector(a, b, 0, 1, 2, 3, 4, 5, 6, 7,
                                         8, 9, 10, 11, 12, 13, 14, 15);
}

__device__ __forceinline__ v8f vzero8() {
    v8f z = {0.f, 0.f, 0.f, 0.f, 0.f, 0.f, 0.f, 0.f};
    return z;
}

// ---------------------------------------------------------------------------
// CDNA5 async global->LDS copy (ASYNCcnt-tracked), 16B per lane.
// Builtin takes pointers to 4xi32 vectors (per hipcc diagnostic).
// Falls back to a plain v8h copy if the builtin is unavailable.
// ---------------------------------------------------------------------------
#if defined(__gfx1250__) && __has_builtin(__builtin_amdgcn_global_load_async_to_lds_b128)
#define HAVE_ASYNC_LDS 1
#endif

__device__ __forceinline__ void async_copy16(const _Float16* g, _Float16* l) {
#ifdef HAVE_ASYNC_LDS
    __builtin_amdgcn_global_load_async_to_lds_b128(
        (v4i*)(void*)g, (v4i*)(void*)l, 0, 0);
#else
    *reinterpret_cast<v8h*>(l) = *reinterpret_cast<const v8h*>(g);
#endif
}

__device__ __forceinline__ void async_wait() {
#ifdef HAVE_ASYNC_LDS
    asm volatile("s_wait_asynccnt 0x0" ::: "memory");
#endif
}

#define NROWS 8192   // B*S
#define DMODEL 512
#define SEQ 1024
#define NHEAD 8
#define DEPTH 64

// ---------------------------------------------------------------------------
// Generic WMMA GEMM: C = act(A[M,K](f16) @ W[K,N](f32) + bias).
// A tiles async-copied to LDS; W tiles converted f32->f16 during staging.
// Tiles 64x64x32, 4 waves; outputs optionally f32 and/or f16.
// M multiple of 64, K multiple of 32; N guarded (handles the N=4 head).
// ---------------------------------------------------------------------------
__global__ __launch_bounds__(128) void k_gemm_wmma(
    const _Float16* __restrict__ A, const float* __restrict__ W,
    const float* __restrict__ bias, float* __restrict__ C32,
    _Float16* __restrict__ C16, int M, int N, int K, int relu)
{
    __shared__ _Float16 As[64 * 40];   // [row][k], stride 40 halves (16B aligned)
    __shared__ _Float16 Bs[64 * 40];   // transposed: [n][k]

    const int tid  = threadIdx.x;
    const int lane = tid & 31;
    const int wv   = tid >> 5;
    const int l    = lane & 15;
    const int sub  = lane >> 4;          // 0: lanes 0-15, 1: lanes 16-31
    const int m0   = blockIdx.y * 64;
    const int n0   = blockIdx.x * 64;

    v8f acc[4];
#pragma unroll
    for (int c = 0; c < 4; ++c) acc[c] = vzero8();

    for (int k0 = 0; k0 < K; k0 += 32) {
        __syncthreads();
        // --- stage A tile 64x32 f16: async copy, 8-half chunks ---
#pragma unroll
        for (int t = 0; t < 2; ++t) {
            int id = t * 128 + tid;          // 256 chunks
            int r  = id >> 2;
            int c  = id & 3;
            async_copy16(&A[(size_t)(m0 + r) * K + k0 + 8 * c],
                         &As[r * 40 + 8 * c]);
        }
        // --- stage B tile 32x64 f32 -> f16 transposed into Bs[n][k] ---
#pragma unroll
        for (int t = 0; t < 4; ++t) {
            int id = t * 128 + tid;          // 512 float4s
            int k  = id >> 4;
            int q  = id & 15;
            int n  = n0 + 4 * q;
            float vx = 0.f, vy = 0.f, vz = 0.f, vw = 0.f;
            if (n + 3 < N) {
                const float4 v = *reinterpret_cast<const float4*>(
                    &W[(size_t)(k0 + k) * N + n]);
                vx = v.x; vy = v.y; vz = v.z; vw = v.w;
            } else {
                if (n     < N) vx = W[(size_t)(k0 + k) * N + n];
                if (n + 1 < N) vy = W[(size_t)(k0 + k) * N + n + 1];
                if (n + 2 < N) vz = W[(size_t)(k0 + k) * N + n + 2];
                if (n + 3 < N) vw = W[(size_t)(k0 + k) * N + n + 3];
            }
            Bs[(4 * q + 0) * 40 + k] = (_Float16)vx;
            Bs[(4 * q + 1) * 40 + k] = (_Float16)vy;
            Bs[(4 * q + 2) * 40 + k] = (_Float16)vz;
            Bs[(4 * q + 3) * 40 + k] = (_Float16)vw;
        }
        async_wait();
        __syncthreads();

        // --- compute: one A-frag per wave, 4 B-frags (N chunks of 16) ---
        const int row = wv * 16 + l;
        v16h a = frag16(&As[row * 40 + sub * 8], &As[row * 40 + 16 + sub * 8]);
#pragma unroll
        for (int c = 0; c < 4; ++c) {
            int n = c * 16 + l;
            v16h b = frag16(&Bs[n * 40 + sub * 8], &Bs[n * 40 + 16 + sub * 8]);
            acc[c] = wmma16(a, b, acc[c]);
        }
    }

    // --- epilogue: bias (+relu), bounds-checked stores ---
#pragma unroll
    for (int c = 0; c < 4; ++c) {
        int col = n0 + c * 16 + l;
        float bv = (col < N) ? bias[col] : 0.f;
#pragma unroll
        for (int r = 0; r < 8; ++r) {
            int row = m0 + wv * 16 + sub * 8 + r;
            float v = acc[c][r] + bv;
            if (relu) v = fmaxf(v, 0.f);
            if (col < N) {
                if (C32) C32[(size_t)row * N + col] = v;
                if (C16) C16[(size_t)row * N + col] = (_Float16)v;
            }
        }
    }
}

// ---------------------------------------------------------------------------
// Flash attention with relative-position skew.
// Q,K,V: f16 [B*S, 512] (head h at cols h*64..); E: f16 [S, 64].
// srel[i,j] = q_i . E[S-1-i+j]  (valid for j<=i; j>i causally masked).
// Block: 4 waves, 64 query rows (16 per wave), one (b,h); streams keys in
// 32-wide tiles with online softmax. O (f16) at head cols of [B*S,512].
// ---------------------------------------------------------------------------
__global__ __launch_bounds__(128) void k_flash_attn(
    const _Float16* __restrict__ Q, const _Float16* __restrict__ Kg,
    const _Float16* __restrict__ V, const _Float16* __restrict__ E,
    _Float16* __restrict__ O)
{
    __shared__ _Float16 Qs[64 * 64];       // [qrow][d]
    __shared__ _Float16 Ks[32 * 64];       // [key][d]
    __shared__ _Float16 Vst[64 * 32];      // transposed: [d][key]
    __shared__ float    QEs[4][16 * 48];   // per-wave skew window [qrow][off]
    __shared__ _Float16 Ps[4][16 * 32];    // per-wave probs [qrow][key]

    const int tid  = threadIdx.x;
    const int lane = tid & 31;
    const int wv   = tid >> 5;
    const int l    = lane & 15;
    const int sub  = lane >> 4;
    const int i0   = blockIdx.x * 64;            // first q row of block
    const int bh   = blockIdx.y;
    const int b    = bh >> 3;
    const int h    = bh & 7;
    const size_t baseQ  = ((size_t)b * SEQ + i0) * DMODEL + h * DEPTH;
    const size_t baseKV = ((size_t)b * SEQ) * DMODEL + h * DEPTH;

    // stage Q tile 64x64 f16 via async copy
#pragma unroll
    for (int t = 0; t < 4; ++t) {
        int id = t * 128 + tid;                  // 512 chunks of 8 halves
        int r  = id >> 3;
        int c  = id & 7;
        async_copy16(&Q[baseQ + (size_t)r * DMODEL + 8 * c], &Qs[r * 64 + 8 * c]);
    }
    async_wait();
    __syncthreads();

    const int rowq = wv * 16 + l;
    v16h aq0 = frag16(&Qs[rowq * 64 +      sub * 8], &Qs[rowq * 64 + 16 + sub * 8]);
    v16h aq1 = frag16(&Qs[rowq * 64 + 32 + sub * 8], &Qs[rowq * 64 + 48 + sub * 8]);

    v8f o0 = vzero8(), o1 = vzero8(), o2 = vzero8(), o3 = vzero8();
    float mrun[8], lsum[8];
#pragma unroll
    for (int r = 0; r < 8; ++r) { mrun[r] = -3.0e38f; lsum[r] = 0.f; }

    const int i0w    = i0 + wv * 16;
    const int nsteps = i0 / 32 + 2;              // uniform per block (causal)

    for (int st = 0; st < nsteps; ++st) {
        const int j0 = st * 32;
        __syncthreads();
        // K tile: async copy (layout identical); V tile: load + transpose
#pragma unroll
        for (int t = 0; t < 2; ++t) {
            int id = t * 128 + tid;              // 256 chunks of 8 halves
            int kr = id >> 3;
            int c  = id & 7;
            async_copy16(&Kg[baseKV + (size_t)(j0 + kr) * DMODEL + 8 * c],
                         &Ks[kr * 64 + 8 * c]);
            v8h vv = *reinterpret_cast<const v8h*>(
                &V[baseKV + (size_t)(j0 + kr) * DMODEL + 8 * c]);
#pragma unroll
            for (int u = 0; u < 8; ++u)
                Vst[(8 * c + u) * 32 + kr] = vv[u];
        }
        async_wait();
        __syncthreads();

        // --- relative-position window: qe = Q_tile @ E_win^T via WMMA,
        //     m = mlo + (15 - M) + n  with off range [0, 47] ---
        const int mlo = (SEQ - 1) - (i0w + 15) + j0;
#pragma unroll
        for (int c = 0; c < 3; ++c) {
            int m = mlo + c * 16 + l;
            m = m < 0 ? 0 : (m > SEQ - 1 ? SEQ - 1 : m);
            const _Float16* ep = &E[(size_t)m * DEPTH];
            v16h e0 = frag16(&ep[sub * 8],      &ep[16 + sub * 8]);
            v16h e1 = frag16(&ep[32 + sub * 8], &ep[48 + sub * 8]);
            v8f w = vzero8();
            w = wmma16(aq0, e0, w);
            w = wmma16(aq1, e1, w);
#pragma unroll
            for (int r = 0; r < 8; ++r)
                QEs[wv][(sub * 8 + r) * 48 + c * 16 + l] = w[r];
        }

        // --- scores = Q @ K^T, add skew band, scale, causal mask ---
        v8f s[2];
#pragma unroll
        for (int c = 0; c < 2; ++c) {
            int key = c * 16 + l;
            v16h b0 = frag16(&Ks[key * 64 +      sub * 8], &Ks[key * 64 + 16 + sub * 8]);
            v16h b1 = frag16(&Ks[key * 64 + 32 + sub * 8], &Ks[key * 64 + 48 + sub * 8]);
            v8f t2 = vzero8();
            t2 = wmma16(aq0, b0, t2);
            t2 = wmma16(aq1, b1, t2);
#pragma unroll
            for (int r = 0; r < 8; ++r) {
                int Mr = sub * 8 + r;
                int nl = c * 16 + l;
                float srel = QEs[wv][Mr * 48 + (15 - Mr) + nl];
                int i = i0w + Mr;
                int j = j0 + nl;
                t2[r] = (t2[r] + srel) * 0.125f + (j > i ? -1.0e9f : 0.f);
            }
            s[c] = t2;
        }

        // --- online softmax; row M spans a 16-lane half, reduce via shfl ---
#pragma unroll
        for (int r = 0; r < 8; ++r) {
            float tm = fmaxf(s[0][r], s[1][r]);
            tm = fmaxf(tm, __shfl_xor(tm, 1));
            tm = fmaxf(tm, __shfl_xor(tm, 2));
            tm = fmaxf(tm, __shfl_xor(tm, 4));
            tm = fmaxf(tm, __shfl_xor(tm, 8));
            float mn = fmaxf(mrun[r], tm);
            float sc = __expf(mrun[r] - mn);
            mrun[r] = mn;
            float p0 = __expf(s[0][r] - mn);
            float p1 = __expf(s[1][r] - mn);
            float rs = p0 + p1;
            rs += __shfl_xor(rs, 1);
            rs += __shfl_xor(rs, 2);
            rs += __shfl_xor(rs, 4);
            rs += __shfl_xor(rs, 8);
            lsum[r] = lsum[r] * sc + rs;
            o0[r] *= sc; o1[r] *= sc; o2[r] *= sc; o3[r] *= sc;
            Ps[wv][(sub * 8 + r) * 32 +      l] = (_Float16)p0;
            Ps[wv][(sub * 8 + r) * 32 + 16 + l] = (_Float16)p1;
        }

        // --- O += P @ V via WMMA (P staged to LDS -> A-fragment layout) ---
        v16h ap = frag16(&Ps[wv][l * 32 + sub * 8], &Ps[wv][l * 32 + 16 + sub * 8]);
#pragma unroll
        for (int c = 0; c < 4; ++c) {
            int d0 = c * 16 + l;
            v16h vb = frag16(&Vst[d0 * 32 + sub * 8], &Vst[d0 * 32 + 16 + sub * 8]);
            v8f* oc = (c == 0) ? &o0 : (c == 1) ? &o1 : (c == 2) ? &o2 : &o3;
            *oc = wmma16(ap, vb, *oc);
        }
    }

    // --- normalize and write out (f16, consumed only by the Wo GEMM) ---
#pragma unroll
    for (int r = 0; r < 8; ++r) {
        float inv = 1.0f / lsum[r];
        int Mr = sub * 8 + r;
        size_t rowo = ((size_t)b * SEQ + i0w + Mr) * DMODEL + h * DEPTH;
        O[rowo +  0 + l] = (_Float16)(o0[r] * inv);
        O[rowo + 16 + l] = (_Float16)(o1[r] * inv);
        O[rowo + 32 + l] = (_Float16)(o2[r] * inv);
        O[rowo + 48 + l] = (_Float16)(o3[r] * inv);
    }
}

// ---------------------------------------------------------------------------
// Embedding gather + concat -> f16 [B*S, 512] (feeds only the input GEMM)
// ---------------------------------------------------------------------------
__global__ __launch_bounds__(256) void k_embed(
    const int* __restrict__ inp, const float* __restrict__ e0,
    const float* __restrict__ e1, const float* __restrict__ e2,
    const float* __restrict__ e3, _Float16* __restrict__ out)
{
    int gid = blockIdx.x * 256 + threadIdx.x;
    int r = gid >> 9;
    int c = gid & 511;
    float v;
    if (c < 64)        v = e0[inp[r * 4 + 0] * 64  + c];
    else if (c < 320)  v = e1[inp[r * 4 + 1] * 256 + (c - 64)];
    else if (c < 448)  v = e2[inp[r * 4 + 2] * 128 + (c - 320)];
    else               v = e3[inp[r * 4 + 3] * 64  + (c - 448)];
    out[gid] = (_Float16)v;
}

// ---------------------------------------------------------------------------
// x = x*sqrt(D) + positional encoding; writes f32 master + f16 mirror
// ---------------------------------------------------------------------------
__global__ __launch_bounds__(256) void k_scale_pos(
    float* __restrict__ x, _Float16* __restrict__ xh)
{
    int gid = blockIdx.x * 256 + threadIdx.x;
    int r = gid >> 9;
    int c = gid & 511;
    int s = r & (SEQ - 1);
    float e = (float)(c & ~1) * (1.0f / 512.0f);
    float angle = (float)s * __powf(10000.0f, -e);
    float pe = (c & 1) ? __cosf(angle) : __sinf(angle);
    float v = x[gid] * 22.62741699796952f + pe;   // sqrt(512)
    x[gid]  = v;
    xh[gid] = (_Float16)v;
}

// ---------------------------------------------------------------------------
// out = LayerNorm(a + b) * g + beta   (rows of 512); f32 + f16 outputs
// ---------------------------------------------------------------------------
__global__ __launch_bounds__(128) void k_add_ln(
    const float* __restrict__ A, const float* __restrict__ Bb,
    const float* __restrict__ g, const float* __restrict__ be,
    float* __restrict__ out, _Float16* __restrict__ outh)
{
    const int row = blockIdx.x;
    const int tid = threadIdx.x;
    __shared__ float red[2][4];

    float v[4];
    float s = 0.f, sq = 0.f;
#pragma unroll
    for (int u = 0; u < 4; ++u) {
        int c = tid + u * 128;
        float x = A[(size_t)row * 512 + c] + Bb[(size_t)row * 512 + c];
        v[u] = x; s += x; sq += x * x;
    }
#pragma unroll
    for (int m = 16; m >= 1; m >>= 1) {
        s  += __shfl_xor(s,  m);
        sq += __shfl_xor(sq, m);
    }
    int wv = tid >> 5, lane = tid & 31;
    if (lane == 0) { red[0][wv] = s; red[1][wv] = sq; }
    __syncthreads();
    s  = red[0][0] + red[0][1] + red[0][2] + red[0][3];
    sq = red[1][0] + red[1][1] + red[1][2] + red[1][3];
    float mean = s * (1.f / 512.f);
    float var  = sq * (1.f / 512.f) - mean * mean;
    float inv  = rsqrtf(var + 1e-6f);
#pragma unroll
    for (int u = 0; u < 4; ++u) {
        int c = tid + u * 128;
        float y = (v[u] - mean) * inv * g[c] + be[c];
        out[(size_t)row * 512 + c]  = y;
        outh[(size_t)row * 512 + c] = (_Float16)y;
    }
}

// ---------------------------------------------------------------------------
// x_ = concat(x, emb0[tar[...,0]]) -> f32 [B*S,576] (model output) + f16 mirror
// ---------------------------------------------------------------------------
__global__ __launch_bounds__(128) void k_concat_xe(
    const float* __restrict__ X, const int* __restrict__ tar,
    const float* __restrict__ e0, float* __restrict__ out,
    _Float16* __restrict__ outh)
{
    int gid = blockIdx.x * 128 + threadIdx.x;
    int r = gid / 576;
    int c = gid - r * 576;
    float v = (c < 512) ? X[(size_t)r * 512 + c]
                        : e0[tar[r * 4 + 0] * 64 + (c - 512)];
    out[gid]  = v;
    outh[gid] = (_Float16)v;
}

// ---------------------------------------------------------------------------
// f32 -> f16 conversion (used for Erel)
// ---------------------------------------------------------------------------
__global__ __launch_bounds__(256) void k_cvt_f16(
    const float* __restrict__ in, _Float16* __restrict__ out, int n)
{
    int i = blockIdx.x * 256 + threadIdx.x;
    if (i < n) out[i] = (_Float16)in[i];
}

// ---------------------------------------------------------------------------
// Host orchestration
// ---------------------------------------------------------------------------
extern "C" void kernel_launch(void* const* d_in, const int* in_sizes, int n_in,
                              void* d_out, int out_size, void* d_ws, size_t ws_size,
                              hipStream_t stream)
{
    (void)in_sizes; (void)n_in; (void)out_size; (void)ws_size;

    const int*   inputs = (const int*)d_in[0];
    const int*   tar    = (const int*)d_in[1];
    const float* emb0   = (const float*)d_in[2];
    const float* emb1   = (const float*)d_in[3];
    const float* emb2   = (const float*)d_in[4];
    const float* emb3   = (const float*)d_in[5];
    const float* Win    = (const float*)d_in[6];
    const float* bin    = (const float*)d_in[7];
    const float* Wq     = (const float*)d_in[8];
    const float* bq     = (const float*)d_in[9];
    const float* Wk     = (const float*)d_in[10];
    const float* bk     = (const float*)d_in[11];
    const float* Wv     = (const float*)d_in[12];
    const float* bv     = (const float*)d_in[13];
    const float* Wo     = (const float*)d_in[14];
    const float* bo     = (const float*)d_in[15];
    const float* Erel   = (const float*)d_in[16];
    const float* ln1g   = (const float*)d_in[17];
    const float* ln1b   = (const float*)d_in[18];
    const float* Wf1    = (const float*)d_in[19];
    const float* bf1    = (const float*)d_in[20];
    const float* Wf2    = (const float*)d_in[21];
    const float* bf2    = (const float*)d_in[22];
    const float* ln2g   = (const float*)d_in[23];
    const float* ln2b   = (const float*)d_in[24];
    const float* Wout0  = (const float*)d_in[25];
    const float* Wout1  = (const float*)d_in[26];
    const float* Wout2  = (const float*)d_in[27];
    const float* Wout3  = (const float*)d_in[28];
    const float* bout0  = (const float*)d_in[29];
    const float* bout1  = (const float*)d_in[30];
    const float* bout2  = (const float*)d_in[31];
    const float* bout3  = (const float*)d_in[32];

    const size_t R = NROWS;
    float* ws = (float*)d_ws;
    size_t fo = 0;
    float* WX  = ws + fo; fo += R * 512;     // f32 residual stream
    float* WX1 = ws + fo; fo += R * 512;     // f32 post-LN1
    float* WT  = ws + fo; fo += R * 512;     // f32 proj / FFN2 scratch

    _Float16* hb = (_Float16*)(ws + fo);
    size_t ho = 0;
    _Float16* WXh  = hb + ho; ho += R * 512;     // f16 mirror of WX
    _Float16* WX1h = hb + ho; ho += R * 512;     // f16 mirror of WX1
    _Float16* WQh  = hb + ho; ho += R * 512;
    _Float16* WKh  = hb + ho; ho += R * 512;
    _Float16* WVh  = hb + ho; ho += R * 512;
    _Float16* WOh  = hb + ho; ho += R * 512;     // flash output
    _Float16* WTh  = hb + ho; ho += R * 512;     // embedding concat
    _Float16* WF1h = hb + ho; ho += R * 2048;    // FFN hidden
    _Float16* xoh  = hb + ho; ho += R * 576;     // x_ mirror
    _Float16* Ef   = hb + ho; ho += (size_t)6 * SEQ * DEPTH;   // Erel f16

    float* out = (float*)d_out;
    float* o0  = out;                 // [8192,4]
    float* o1  = o0 + R * 4;          // [8192,256]
    float* o2  = o1 + R * 256;        // [8192,128]
    float* o3  = o2 + R * 128;        // [8192,64]
    float* xo  = o3 + R * 64;         // x_ [8192,576]

    const dim3 blk(128);
    auto gemm = [&](const _Float16* A, const float* W, const float* bias,
                    float* C32, _Float16* C16, int M, int N, int K, int relu) {
        dim3 g((unsigned)((N + 63) / 64), (unsigned)(M / 64));
        k_gemm_wmma<<<g, blk, 0, stream>>>(A, W, bias, C32, C16, M, N, K, relu);
    };

    // Erel -> f16 once per call
    k_cvt_f16<<<(6 * SEQ * DEPTH) / 256, 256, 0, stream>>>(Erel, Ef, 6 * SEQ * DEPTH);

    // embed -> input projection -> scale + positional encoding
    k_embed<<<NROWS * 512 / 256, 256, 0, stream>>>(inputs, emb0, emb1, emb2, emb3, WTh);
    gemm(WTh, Win, bin, WX, nullptr, NROWS, 512, 512, 0);
    k_scale_pos<<<NROWS * 512 / 256, 256, 0, stream>>>(WX, WXh);

    for (int lr = 0; lr < 6; ++lr) {
        const size_t wofs = (size_t)lr * 512 * 512;
        gemm(WXh, Wq + wofs, bq + lr * 512, nullptr, WQh, NROWS, 512, 512, 0);
        gemm(WXh, Wk + wofs, bk + lr * 512, nullptr, WKh, NROWS, 512, 512, 0);
        gemm(WXh, Wv + wofs, bv + lr * 512, nullptr, WVh, NROWS, 512, 512, 0);

        k_flash_attn<<<dim3(SEQ / 64, 8 * NHEAD), blk, 0, stream>>>(
            WQh, WKh, WVh, Ef + (size_t)lr * SEQ * DEPTH, WOh);

        gemm(WOh, Wo + wofs, bo + lr * 512, WT, nullptr, NROWS, 512, 512, 0);
        k_add_ln<<<NROWS, blk, 0, stream>>>(WX, WT, ln1g + lr * 512, ln1b + lr * 512,
                                            WX1, WX1h);

        gemm(WX1h, Wf1 + (size_t)lr * 512 * 2048, bf1 + lr * 2048,
             nullptr, WF1h, NROWS, 2048, 512, 1);
        gemm(WF1h, Wf2 + (size_t)lr * 2048 * 512, bf2 + lr * 512,
             WT, nullptr, NROWS, 512, 2048, 0);
        k_add_ln<<<NROWS, blk, 0, stream>>>(WX1, WT, ln2g + lr * 512, ln2b + lr * 512,
                                            WX, WXh);
    }

    // heads
    gemm(WXh, Wout0, bout0, o0, nullptr, NROWS, 4, 512, 0);
    k_concat_xe<<<NROWS * 576 / 128, blk, 0, stream>>>(WX, tar, emb0, xo, xoh);
    gemm(xoh, Wout1, bout1, o1, nullptr, NROWS, 256, 576, 0);
    gemm(xoh, Wout2, bout2, o2, nullptr, NROWS, 128, 576, 0);
    gemm(xoh, Wout3, bout3, o3, nullptr, NROWS, 64, 576, 0);
}
